// GNN_12884901888248
// MI455X (gfx1250) — matrix operand order repaired and checked by reading, not compiled, not run on hardware
//
#include <hip/hip_runtime.h>
#include <hip/hip_bf16.h>
#include <math.h>

// ---------------------------------------------------------------------------
// GNN message passing on MI455X (gfx1250, wave32, WMMA).
//
// Per iteration (x5):
//   x[E,80] = [edge_feat | state[edge_dst]]  ->  tanh(x@Wst1+b)  [E,32]
//   -> tanh(h1@Wst2+b) [E,16] -> atomic segment_sum by edge_src -> new_state
//   freeze logic: done[i+1] = done[i] | all(||new-old||<=0.01)
// Final: softmax(tanh(state@Wout1+b)@Wout2+b) per node.
//
// WMMA mapping: 16 edges = M tile. Layer1: K=80 pad 96 (3 chunks x32) x
// N=32 (2 tiles) -> 6x v_wmma_f32_16x16x32_f16. LDS transpose (D->A layout),
// then 1 WMMA for layer 2.  f16 inputs / f32 accumulate.
// tanh uses the native gfx1250 V_TANH_F32 TRANS op (co-executes with XDL).
// ---------------------------------------------------------------------------

typedef __attribute__((ext_vector_type(16))) _Float16 v16h;
typedef __attribute__((ext_vector_type(8)))  float    v8f;

#define STATE_DIM 16
#define MAX_IT 5
#define WAVES_PER_BLOCK 8
#define LDS_STRIDE 33   // 16x32 f32 tile, padded to dodge bank conflicts

// Branch-free tanh: native V_TANH_F32 when the builtin exists, else a
// rational exp2 form (single v_exp_f32 + v_rcp_f32, no exec-mask churn).
__device__ __forceinline__ float fast_tanh(float x) {
#if __has_builtin(__builtin_amdgcn_tanhf)
    return __builtin_amdgcn_tanhf(x);
#elif __has_builtin(__builtin_amdgcn_tanh_f32)
    return __builtin_amdgcn_tanh_f32(x);
#else
    // tanh(x) = 1 - 2/(exp2(2*log2e*x) + 1); saturates correctly at +-1
    float e2 = __builtin_amdgcn_exp2f(x * 2.8853900817779268f);
    return 1.0f - 2.0f * __builtin_amdgcn_rcpf(e2 + 1.0f);
#endif
}

__device__ __forceinline__ void put4(v16h& v, int base, float4 q) {
    v[base + 0] = (_Float16)q.x;
    v[base + 1] = (_Float16)q.y;
    v[base + 2] = (_Float16)q.z;
    v[base + 3] = (_Float16)q.w;
}

// ---------------------------------------------------------------------------
// Zero a float buffer + reset one int flag (new_state / nonconv per iteration,
// also used for initial state0 / done[0]).
// ---------------------------------------------------------------------------
__global__ void gnn_zero_kernel(float* __restrict__ p, int n, int* __restrict__ flag) {
    int i = blockIdx.x * blockDim.x + threadIdx.x;
    if (i == 0 && flag) *flag = 0;
    int stride = gridDim.x * blockDim.x;
    for (; i < n; i += stride) p[i] = 0.0f;
}

// ---------------------------------------------------------------------------
// Edge kernel: one wave processes 16-edge tiles (grid-stride). Emits the
// 7 WMMAs per tile and scatters the result with global_atomic_add_f32.
// ---------------------------------------------------------------------------
__global__ __launch_bounds__(256) void gnn_edge_kernel(
    const float* __restrict__ edge_feat,   // [E,64]
    const float* __restrict__ Wst1,        // [80,32]
    const float* __restrict__ bst1,        // [32]
    const float* __restrict__ Wst2,        // [32,16]
    const float* __restrict__ bst2,        // [16]
    const int*   __restrict__ edge_src,
    const int*   __restrict__ edge_dst,
    const float* __restrict__ state,       // [N,16]
    float*       __restrict__ new_state,   // [N,16] (pre-zeroed)
    const int*   __restrict__ done_flag,
    int n_edges)
{
    if (*done_flag) return;   // frozen: message pass is dead work

    __shared__ float lds_h1[WAVES_PER_BLOCK][16 * LDS_STRIDE];

    const int lane = threadIdx.x & 31;
    const int wave = threadIdx.x >> 5;
    const int row  = lane & 15;   // M row (edge within tile) / N col for B,C,D
    const int kh   = lane >> 4;   // which K-half this lane owns

    // ---- Build f16 B operands once per wave (weights are tiny, L2-hot). ----
    // B layout (16-bit, 32x16): lane half h holds column N=lane&15;
    // elem j <-> K=8h+j (j<8), K=16+8h+(j-8) (j>=8), matching A.
    v16h B1[3][2];   // [K-chunk][N-tile] for Wst1 (K padded 80->96)
#pragma unroll
    for (int c = 0; c < 3; ++c)
#pragma unroll
        for (int t = 0; t < 2; ++t) {
            int col = t * 16 + row;
#pragma unroll
            for (int j = 0; j < 16; ++j) {
                int k = c * 32 + (j < 8 ? kh * 8 + j : 16 + kh * 8 + (j - 8));
                float w = (k < 80) ? Wst1[k * 32 + col] : 0.0f;
                B1[c][t][j] = (_Float16)w;
            }
        }
    v16h B2;         // Wst2 [32,16]
#pragma unroll
    for (int j = 0; j < 16; ++j) {
        int k = (j < 8 ? kh * 8 + j : 16 + kh * 8 + (j - 8));
        B2[j] = (_Float16)Wst2[k * 16 + row];
    }
    const float b1a = bst1[row];        // bias for N-tile 0 (col = lane&15)
    const float b1b = bst1[16 + row];   // bias for N-tile 1
    const float b2  = bst2[row];

    const int n_tiles = (n_edges + 15) >> 4;
    const int wave_id = blockIdx.x * WAVES_PER_BLOCK + wave;
    const int n_waves = gridDim.x * WAVES_PER_BLOCK;
    float* L = lds_h1[wave];

    for (int T = wave_id; T < n_tiles; T += n_waves) {
        // ---- Build A operands (16x32 f16 each). Lane owns row = lane&15. ----
        int e  = T * 16 + row;
        int es = (e < n_edges) ? e : 0;
        int dst = edge_dst[es];
        const float4* f4 = reinterpret_cast<const float4*>(edge_feat + (size_t)es * 64);
        const float4* s4 = reinterpret_cast<const float4*>(state + (size_t)dst * STATE_DIM);

        v16h A0, A1, A2;
        // chunk 0: K 0..31 of x  (edge_feat cols 0..31)
        put4(A0, 0,  f4[kh * 2 + 0]);       // K = kh*8 + 0..3
        put4(A0, 4,  f4[kh * 2 + 1]);       // K = kh*8 + 4..7
        put4(A0, 8,  f4[4 + kh * 2 + 0]);   // K = 16 + kh*8 + 0..3
        put4(A0, 12, f4[4 + kh * 2 + 1]);
        // chunk 1: K 32..63 (edge_feat cols 32..63)
        put4(A1, 0,  f4[8 + kh * 2 + 0]);
        put4(A1, 4,  f4[8 + kh * 2 + 1]);
        put4(A1, 8,  f4[12 + kh * 2 + 0]);
        put4(A1, 12, f4[12 + kh * 2 + 1]);
        // chunk 2: K 64..79 = gathered state, K 80..95 = zero pad
        put4(A2, 0,  s4[kh * 2 + 0]);
        put4(A2, 4,  s4[kh * 2 + 1]);
#pragma unroll
        for (int j = 8; j < 16; ++j) A2[j] = (_Float16)0.0f;

        // ---- GEMM1: x[16x96] @ Wst1[96x32] -> two 16x16 f32 tiles ----
        v8f acc0 = {}, acc1 = {};
        acc0 = __builtin_amdgcn_wmma_f32_16x16x32_f16(false, A0, false, B1[0][0], (short)0, acc0, false, false);
        acc0 = __builtin_amdgcn_wmma_f32_16x16x32_f16(false, A1, false, B1[1][0], (short)0, acc0, false, false);
        acc0 = __builtin_amdgcn_wmma_f32_16x16x32_f16(false, A2, false, B1[2][0], (short)0, acc0, false, false);
        acc1 = __builtin_amdgcn_wmma_f32_16x16x32_f16(false, A0, false, B1[0][1], (short)0, acc1, false, false);
        acc1 = __builtin_amdgcn_wmma_f32_16x16x32_f16(false, A1, false, B1[1][1], (short)0, acc1, false, false);
        acc1 = __builtin_amdgcn_wmma_f32_16x16x32_f16(false, A2, false, B1[2][1], (short)0, acc1, false, false);

        // ---- bias + tanh (native V_TANH_F32), D-layout -> LDS ----
        // D layout: lane l, VGPR v holds D[v + 8*(l>>4)][l&15].
#pragma unroll
        for (int v = 0; v < 8; ++v) {
            int M = v + 8 * kh;
            L[M * LDS_STRIDE + row]      = fast_tanh(acc0[v] + b1a);
            L[M * LDS_STRIDE + 16 + row] = fast_tanh(acc1[v] + b1b);
        }
        asm volatile("s_wait_dscnt 0" ::: "memory");  // intra-wave LDS RAW

        // ---- Re-read h1 in A layout (f16) for GEMM2 ----
        v16h Ah;
#pragma unroll
        for (int j = 0; j < 8; ++j) {
            Ah[j]     = (_Float16)L[row * LDS_STRIDE + kh * 8 + j];
            Ah[8 + j] = (_Float16)L[row * LDS_STRIDE + 16 + kh * 8 + j];
        }

        // ---- GEMM2: h1[16x32] @ Wst2[32x16] ----
        v8f acc2 = {};
        acc2 = __builtin_amdgcn_wmma_f32_16x16x32_f16(false, Ah, false, B2, (short)0, acc2, false, false);

        // ---- bias + tanh + scatter (segment_sum via f32 atomics) ----
#pragma unroll
        for (int v = 0; v < 8; ++v) {
            int M  = v + 8 * kh;
            int ee = T * 16 + M;
            if (ee < n_edges) {
                float r = fast_tanh(acc2[v] + b2);
                int node = edge_src[ee];
                unsafeAtomicAdd(&new_state[(size_t)node * STATE_DIM + row], r);
            }
        }
        asm volatile("s_wait_dscnt 0" ::: "memory");  // protect LDS tile reuse
    }
}

// ---------------------------------------------------------------------------
// Node kernel: per-node convergence distance + freeze select, ping-pong write.
// ---------------------------------------------------------------------------
__global__ __launch_bounds__(256) void gnn_node_kernel(
    const float* __restrict__ cur, const float* __restrict__ newst,
    float* __restrict__ nxt, const int* __restrict__ done_in,
    int* __restrict__ nonconv, int n_nodes)
{
    int node = blockIdx.x * blockDim.x + threadIdx.x;
    if (node >= n_nodes) return;
    const int done = *done_in;
    const float4* o4 = reinterpret_cast<const float4*>(cur   + (size_t)node * STATE_DIM);
    const float4* n4 = reinterpret_cast<const float4*>(newst + (size_t)node * STATE_DIM);
    float4*       x4 = reinterpret_cast<float4*>(nxt + (size_t)node * STATE_DIM);
    float d2 = 0.0f;
#pragma unroll
    for (int q = 0; q < 4; ++q) {
        float4 o = o4[q], n = n4[q];
        float dx = n.x - o.x, dy = n.y - o.y, dz = n.z - o.z, dw = n.w - o.w;
        d2 += dx * dx + dy * dy + dz * dz + dw * dw;
        x4[q] = done ? o : n;
    }
    if (sqrtf(d2 + 1e-10f) > 0.01f) atomicOr(nonconv, 1);
}

__global__ void gnn_finalize_kernel(const int* __restrict__ done_in,
                                    const int* __restrict__ nonconv,
                                    int* __restrict__ done_out) {
    *done_out = *done_in | (*nonconv == 0);
}

// ---------------------------------------------------------------------------
// Readout: tanh(state@Wout1+b) @ Wout2 + b -> softmax.  ~51 MFLOP total
// (0.4% of the edge MLP), so plain VALU with LDS-staged weights is fine.
// ---------------------------------------------------------------------------
__global__ __launch_bounds__(256) void gnn_out_kernel(
    const float* __restrict__ state,
    const float* __restrict__ Wout1, const float* __restrict__ bout1,
    const float* __restrict__ Wout2, const float* __restrict__ bout2,
    float* __restrict__ out, int n_nodes)
{
    __shared__ float W1[16 * 32], W2[32 * 16], b1[32], b2[16];
    for (int i = threadIdx.x; i < 512; i += blockDim.x) { W1[i] = Wout1[i]; W2[i] = Wout2[i]; }
    if (threadIdx.x < 32) b1[threadIdx.x] = bout1[threadIdx.x];
    if (threadIdx.x < 16) b2[threadIdx.x] = bout2[threadIdx.x];
    __syncthreads();

    int node = blockIdx.x * blockDim.x + threadIdx.x;
    if (node >= n_nodes) return;
    float s[16], h[32], lg[16];
#pragma unroll
    for (int k = 0; k < 16; ++k) s[k] = state[(size_t)node * STATE_DIM + k];
#pragma unroll
    for (int n = 0; n < 32; ++n) {
        float a = b1[n];
#pragma unroll
        for (int k = 0; k < 16; ++k) a = fmaf(s[k], W1[k * 32 + n], a);
        h[n] = fast_tanh(a);
    }
    float mx = -1e30f;
#pragma unroll
    for (int n = 0; n < 16; ++n) {
        float a = b2[n];
#pragma unroll
        for (int k = 0; k < 32; ++k) a = fmaf(h[k], W2[k * 16 + n], a);
        lg[n] = a;
        mx = fmaxf(mx, a);
    }
    float sum = 0.0f;
#pragma unroll
    for (int n = 0; n < 16; ++n) { lg[n] = __expf(lg[n] - mx); sum += lg[n]; }
    float inv = 1.0f / sum;
#pragma unroll
    for (int n = 0; n < 16; ++n) out[(size_t)node * 16 + n] = lg[n] * inv;
}

// ---------------------------------------------------------------------------
extern "C" void kernel_launch(void* const* d_in, const int* in_sizes, int n_in,
                              void* d_out, int out_size, void* d_ws, size_t ws_size,
                              hipStream_t stream) {
    const float* edge_feat = (const float*)d_in[0];
    const float* Wst1  = (const float*)d_in[1];
    const float* bst1  = (const float*)d_in[2];
    const float* Wst2  = (const float*)d_in[3];
    const float* bst2  = (const float*)d_in[4];
    const float* Wout1 = (const float*)d_in[5];
    const float* bout1 = (const float*)d_in[6];
    const float* Wout2 = (const float*)d_in[7];
    const float* bout2 = (const float*)d_in[8];
    const int* edge_src = (const int*)d_in[9];
    const int* edge_dst = (const int*)d_in[10];

    const int n_edges = in_sizes[9];
    const int n_nodes = out_size / 16;   // device scalar unreadable mid-capture
    const size_t nstate = (size_t)n_nodes * STATE_DIM;

    // workspace layout: state_a | state_b | new_state | nonconv | done[0..5]
    float* state_a   = (float*)d_ws;
    float* state_b   = state_a + nstate;
    float* new_state = state_b + nstate;
    int*   nonconv   = (int*)(new_state + nstate);
    int*   done      = nonconv + 1;

    const int zgrid = (int)((nstate + 255) / 256);
    const int ngrid = (n_nodes + 255) / 256;
    const int egrid = 640;   // 5120 waves, ~5 edge-tiles each; weight-load amortized

    // state0 = 0, done[0] = 0
    gnn_zero_kernel<<<zgrid, 256, 0, stream>>>(state_a, (int)nstate, &done[0]);

    float* cur = state_a;
    float* nxt = state_b;
    for (int it = 0; it < MAX_IT; ++it) {
        gnn_zero_kernel<<<zgrid, 256, 0, stream>>>(new_state, (int)nstate, nonconv);
        gnn_edge_kernel<<<egrid, 256, 0, stream>>>(
            edge_feat, Wst1, bst1, Wst2, bst2, edge_src, edge_dst,
            cur, new_state, &done[it], n_edges);
        gnn_node_kernel<<<ngrid, 256, 0, stream>>>(
            cur, new_state, nxt, &done[it], nonconv, n_nodes);
        gnn_finalize_kernel<<<1, 1, 0, stream>>>(&done[it], nonconv, &done[it + 1]);
        float* t = cur; cur = nxt; nxt = t;
    }

    gnn_out_kernel<<<ngrid, 256, 0, stream>>>(
        cur, Wout1, bout1, Wout2, bout2, (float*)d_out, n_nodes);
}
